// BridgeLayer_17265768530808
// MI455X (gfx1250) — matrix operand order repaired and checked by reading, not compiled, not run on hardware
//
#include <hip/hip_runtime.h>

// ---------------------------------------------------------------------------
// BridgeLayer for MI455X (gfx1250, wave32, WMMA).
// All GEMMs run through v_wmma_f32_16x16x32_bf16 (fp32 accumulate).
// Big GEMMs use 32x64 per-wave tiles (8 WMMA per 6 fragment loads).
// Requires ws_size >= ~420 MB (all scratch fully rewritten per call).
// ---------------------------------------------------------------------------

typedef __attribute__((ext_vector_type(16))) __bf16 v16bf;
typedef __attribute__((ext_vector_type(8)))  float  v8f;

__device__ __forceinline__ __bf16 f2bf(float f) {
  union { float f; unsigned u; } v; v.f = f;
  unsigned r = (v.u + 0x7FFFu + ((v.u >> 16) & 1u)) >> 16;
  union { unsigned short s; __bf16 b; } o; o.s = (unsigned short)r; return o.b;
}
__device__ __forceinline__ float bf2f(__bf16 b) {
  union { unsigned short s; __bf16 b; } i; i.b = b;
  union { unsigned u; float f; } o; o.u = ((unsigned)i.s) << 16; return o.f;
}

// ---------------------------------------------------------------------------
// Weight cast + transpose: W[In][Out] fp32 -> Wt[Out][In] bf16, so GEMM B
// fragments read K-contiguous pairs (dword / b128 loads).
// ---------------------------------------------------------------------------
__global__ void wcast_kernel(const float* __restrict__ W, __bf16* __restrict__ Wt,
                             int In, int Out) {
  long long idx = (long long)blockIdx.x * blockDim.x + threadIdx.x;
  long long total = (long long)In * Out;
  if (idx >= total) return;
  int k = (int)(idx / Out);
  int n = (int)(idx % Out);
  Wt[(long long)n * In + k] = f2bf(W[idx]);
}

// ---------------------------------------------------------------------------
// RMSNorm: one 256-thread block per row, output bf16.
// ---------------------------------------------------------------------------
__global__ void rmsnorm_kernel(const float* __restrict__ x, const float* __restrict__ w,
                               __bf16* __restrict__ out, int D) {
  long long row = blockIdx.x;
  const float* xr = x + row * D;
  __bf16* orow = out + row * D;
  float ss = 0.f;
  for (int i = threadIdx.x; i < D; i += 256) { float v = xr[i]; ss += v * v; }
  __shared__ float red[256];
  red[threadIdx.x] = ss; __syncthreads();
  for (int s = 128; s > 0; s >>= 1) {
    if ((int)threadIdx.x < s) red[threadIdx.x] += red[threadIdx.x + s];
    __syncthreads();
  }
  float inv = rsqrtf(red[0] / (float)D + 1e-6f);
  for (int i = threadIdx.x; i < D; i += 256) orow[i] = f2bf(xr[i] * inv * w[i]);
}

// ---------------------------------------------------------------------------
// WMMA GEMM: C[M,Ncols] = A[M,K] @ Bt^T (+ resid) with optional fp32 A
// (converted in-register), bf16/f32 output, transposed store.
// Bt is stored [Ncols][K] (K-contiguous per output column).
// Block: 256 threads = 8 waves. Each wave: (MT*16) x 64 tile; block:
// (MT*128) x 64. B fragments are reused across the MT M-sub-tiles.
// Prefetch of the next K-step is unconditional: it lowers to an SE-scope
// *speculative* global_prefetch_b8, which the ISA drops silently on
// translation failure, so no end-of-buffer guard (and no branches) needed.
// z-dim batching: offsets = (z/Hdiv)*s?b + (z%Hdiv)*s?h (element units).
// ---------------------------------------------------------------------------
template<int MT, bool A_F32, bool OUT_BF16, bool OUT_TRANS, bool RESID>
__global__ void __launch_bounds__(256)
gemm_wmma_kernel(const void* __restrict__ Ap, int lda, long long sAb, long long sAh,
                 const __bf16* __restrict__ Bt, int ldb, long long sBb, long long sBh,
                 void* __restrict__ Cp, int ldc, long long sCb, long long sCh,
                 const float* __restrict__ Rp, long long sRb, long long sRh,
                 int K, int Hdiv, float scale) {
  const int lane = threadIdx.x & 31;
  const int wave = threadIdx.x >> 5;
  const int g    = lane >> 4;   // half-wave select
  const int l16  = lane & 15;
  const int zb = blockIdx.z / Hdiv;
  const int zh = blockIdx.z % Hdiv;
  const int m0 = blockIdx.y * (128 * MT) + wave * (16 * MT);
  const int n0 = blockIdx.x * 64;

  const float*  Af = (const float*)Ap;   // element-unit strides per template flavor
  const __bf16* Ab = (const __bf16*)Ap;
  const __bf16* B  = Bt + zb * sBb + zh * sBh;

  v8f acc[MT][4];
  for (int t = 0; t < MT; t++)
    for (int j = 0; j < 4; j++)
      for (int e = 0; e < 8; e++) acc[t][j][e] = 0.f;

  for (int k0 = 0; k0 < K; k0 += 32) {
    // A fragments: 16x32 bf16, ISA layout (lanes0-15: K 0..7,16..23; +8 for hi half)
    v16bf af[MT];
    #pragma unroll
    for (int t = 0; t < MT; t++) {
      const int mrow = m0 + t * 16 + l16;
      if (A_F32) {
        const float* ar = Af + zb * sAb + zh * sAh + (long long)mrow * lda;
        #pragma unroll
        for (int i = 0; i < 8; i++) {
          int kk = k0 + ((i & 3) << 1) + ((i >> 2) << 4) + (g << 3);
          af[t][2 * i]     = f2bf(ar[kk]);
          af[t][2 * i + 1] = f2bf(ar[kk + 1]);
        }
        __builtin_prefetch(ar + k0 + 32 + (g << 3), 0, 3); // speculative, no guard
      } else {
        const __bf16* ar = Ab + zb * sAb + zh * sAh + (long long)mrow * lda;
        #pragma unroll
        for (int i = 0; i < 8; i++) {
          int kk = k0 + ((i & 3) << 1) + ((i >> 2) << 4) + (g << 3);
          af[t][2 * i]     = ar[kk];
          af[t][2 * i + 1] = ar[kk + 1];
        }
        __builtin_prefetch(ar + k0 + 32 + (g << 3), 0, 3); // speculative, no guard
      }
    }
    // B fragments: 32x16 bf16 each (lanes0-15: K 0..7,16..23; +8 for hi half),
    // each reused across all MT M-sub-tiles.
    #pragma unroll
    for (int j = 0; j < 4; j++) {
      const __bf16* br = B + (long long)(n0 + j * 16 + l16) * ldb;
      v16bf bfv;
      #pragma unroll
      for (int i = 0; i < 8; i++) {
        int kk = k0 + ((i & 3) << 1) + ((i >> 2) << 4) + (g << 3);
        bfv[2 * i]     = br[kk];
        bfv[2 * i + 1] = br[kk + 1];
      }
      #pragma unroll
      for (int t = 0; t < MT; t++) {
        acc[t][j] = __builtin_amdgcn_wmma_f32_16x16x32_bf16(
            false, af[t], false, bfv, (short)0, acc[t][j], false, false);
      }
    }
  }

  // Epilogue: C/D layout — VGPR r holds M=r (lanes0-15) / M=r+8 (lanes16-31)
  #pragma unroll
  for (int t = 0; t < MT; t++) {
    #pragma unroll
    for (int j = 0; j < 4; j++) {
      #pragma unroll
      for (int rr = 0; rr < 8; rr++) {
        int mm = m0 + t * 16 + rr + (g << 3);
        int nn = n0 + j * 16 + l16;
        float v = acc[t][j][rr] * scale;
        if (RESID) v += (Rp + zb * sRb + zh * sRh)[(long long)mm * ldc + nn];
        if (OUT_TRANS) {
          (((__bf16*)Cp) + zb * sCb + zh * sCh)[(long long)nn * ldc + mm] = f2bf(v);
        } else if (OUT_BF16) {
          (((__bf16*)Cp) + zb * sCb + zh * sCh)[(long long)mm * ldc + nn] = f2bf(v);
        } else {
          (((float*)Cp) + zb * sCb + zh * sCh)[(long long)mm * ldc + nn] = v;
        }
      }
    }
  }
}

// ---------------------------------------------------------------------------
// Softmax over rows of length 128 (one wave per row), in place.
// ---------------------------------------------------------------------------
__global__ void softmax128_kernel(float* __restrict__ sc, long long nrows) {
  long long w = (((long long)blockIdx.x * blockDim.x) + threadIdx.x) >> 5;
  if (w >= nrows) return;
  int lane = threadIdx.x & 31;
  float* row = sc + w * 128;
  float v[4];
  float m = -3.4e38f;
  #pragma unroll
  for (int i = 0; i < 4; i++) { v[i] = row[lane + 32 * i]; m = fmaxf(m, v[i]); }
  for (int d = 16; d; d >>= 1) m = fmaxf(m, __shfl_xor(m, d, 32));
  float s = 0.f;
  #pragma unroll
  for (int i = 0; i < 4; i++) { v[i] = __expf(v[i] - m); s += v[i]; }
  for (int d = 16; d; d >>= 1) s += __shfl_xor(s, d, 32);
  float r = 1.f / s;
  #pragma unroll
  for (int i = 0; i < 4; i++) row[lane + 32 * i] = v[i] * r;
}

// ---------------------------------------------------------------------------
// Softmax over rows of length 4096 (one 256-thread block per row), in place.
// ---------------------------------------------------------------------------
__global__ void softmax4096_kernel(float* __restrict__ sc) {
  long long row = blockIdx.x;
  float* p = sc + row * 4096;
  float v[16];
  float m = -3.4e38f;
  #pragma unroll
  for (int i = 0; i < 16; i++) { v[i] = p[threadIdx.x + 256 * i]; m = fmaxf(m, v[i]); }
  __shared__ float red[256];
  red[threadIdx.x] = m; __syncthreads();
  for (int s = 128; s > 0; s >>= 1) {
    if ((int)threadIdx.x < s) red[threadIdx.x] = fmaxf(red[threadIdx.x], red[threadIdx.x + s]);
    __syncthreads();
  }
  m = red[0]; __syncthreads();
  float sum = 0.f;
  #pragma unroll
  for (int i = 0; i < 16; i++) { v[i] = __expf(v[i] - m); sum += v[i]; }
  red[threadIdx.x] = sum; __syncthreads();
  for (int s = 128; s > 0; s >>= 1) {
    if ((int)threadIdx.x < s) red[threadIdx.x] += red[threadIdx.x + s];
    __syncthreads();
  }
  float r = 1.f / red[0];
  #pragma unroll
  for (int i = 0; i < 16; i++) p[threadIdx.x + 256 * i] = v[i] * r;
}

// ---------------------------------------------------------------------------
// Sparse top-8 over N=128, softmax, bf16 gather of v_s rows (Ds=1024),
// s_out += gate * acc. One wave per (b,t).
// ---------------------------------------------------------------------------
__global__ void sparse_topk_kernel(const float* __restrict__ sc2, const __bf16* __restrict__ vs,
                                   const float* __restrict__ gate, float* __restrict__ sout) {
  long long w = (((long long)blockIdx.x * blockDim.x) + threadIdx.x) >> 5; // b*T + t
  int lane = threadIdx.x & 31;
  const float* row = sc2 + w * 128;
  float val[4]; int idx4[4];
  #pragma unroll
  for (int i = 0; i < 4; i++) { idx4[i] = lane + 32 * i; val[i] = row[idx4[i]]; }
  float tv[8]; int ti[8];
  for (int k = 0; k < 8; k++) {
    float m = -3.4e38f; int mi = 0x7fffffff;
    #pragma unroll
    for (int i = 0; i < 4; i++)
      if (val[i] > m || (val[i] == m && idx4[i] < mi)) { m = val[i]; mi = idx4[i]; }
    for (int d = 16; d; d >>= 1) {
      float om = __shfl_xor(m, d, 32);
      int   oi = __shfl_xor(mi, d, 32);
      if (om > m || (om == m && oi < mi)) { m = om; mi = oi; }
    }
    tv[k] = m; ti[k] = mi;
    #pragma unroll
    for (int i = 0; i < 4; i++) if (idx4[i] == mi) val[i] = -3.4e38f;
  }
  float mx = tv[0], s = 0.f, wk[8];
  #pragma unroll
  for (int k = 0; k < 8; k++) { wk[k] = __expf(tv[k] - mx); s += wk[k]; }
  float rinv = 1.f / s;
  long long b = w >> 12; // / T=4096
  float acc[32];
  #pragma unroll
  for (int c = 0; c < 32; c++) acc[c] = 0.f;
  for (int k = 0; k < 8; k++) {
    const __bf16* vrow = vs + ((b * 128 + ti[k]) * 1024);
    float ww = wk[k] * rinv;
    #pragma unroll
    for (int c = 0; c < 32; c++) acc[c] += ww * bf2f(vrow[lane + 32 * c]);
  }
  float g = gate[w];
  float* out = sout + w * 1024;
  #pragma unroll
  for (int c = 0; c < 32; c++) out[lane + 32 * c] += g * acc[c];
}

// ---------------------------------------------------------------------------
extern "C" void kernel_launch(void* const* d_in, const int* in_sizes, int n_in,
                              void* d_out, int out_size, void* d_ws, size_t ws_size,
                              hipStream_t stream) {
  (void)in_sizes; (void)n_in; (void)out_size; (void)ws_size;
  const int B = 8, T = 4096, Nn = 128, Ds = 1024, Dr = 512, H = 8, dhs = 128, dhr = 64;
  const float scale_s = 0.08838834764831845f; // 128^-0.5 (also used for sc2)
  const float scale_r = 0.125f;               // 64^-0.5

  const float* s    = (const float*)d_in[0];
  const float* r    = (const float*)d_in[1];
  const float* gate = (const float*)d_in[2];
  const float* nsw  = (const float*)d_in[3];
  const float* nrw  = (const float*)d_in[4];
  const float* sq   = (const float*)d_in[5];
  const float* sk   = (const float*)d_in[6];
  const float* sv   = (const float*)d_in[7];
  const float* so   = (const float*)d_in[8];
  const float* rq   = (const float*)d_in[9];
  const float* rk   = (const float*)d_in[10];
  const float* rv   = (const float*)d_in[11];
  const float* ro   = (const float*)d_in[12];

  float* s_out = (float*)d_out;
  float* r_out = s_out + (size_t)B * T * Ds;

  char* wsb = (char*)d_ws;
  size_t off = 0;
  auto alloc = [&](size_t bytes) -> char* {
    char* p = wsb + off; off += (bytes + 255) & ~(size_t)255; return p;
  };
  __bf16* sn  = (__bf16*)alloc((size_t)B * T * Ds * 2);
  __bf16* rn  = (__bf16*)alloc((size_t)B * Nn * Dr * 2);
  __bf16* qs  = (__bf16*)alloc((size_t)B * T * Ds * 2);
  __bf16* ks  = (__bf16*)alloc((size_t)B * Nn * Ds * 2);
  __bf16* vs  = (__bf16*)alloc((size_t)B * Nn * Ds * 2);
  __bf16* vts = (__bf16*)alloc((size_t)B * Ds * Nn * 2);  // v_s^T per batch
  __bf16* qrp = (__bf16*)alloc((size_t)B * Nn * Dr * 2);
  __bf16* krp = (__bf16*)alloc((size_t)B * T * Dr * 2);
  __bf16* vrt = (__bf16*)alloc((size_t)B * Dr * T * 2);   // v_r^T per batch
  __bf16* a1  = (__bf16*)alloc((size_t)B * T * Ds * 2);   // attn1 pre-proj
  __bf16* a2  = (__bf16*)alloc((size_t)B * Nn * Dr * 2);  // attn2 pre-proj
  float*  sc  = (float*) alloc((size_t)B * H * T * Nn * 4); // shared score scratch
  __bf16* sqT = (__bf16*)alloc((size_t)Ds * Ds * 2);
  __bf16* skT = (__bf16*)alloc((size_t)Dr * Ds * 2);
  __bf16* svT = (__bf16*)alloc((size_t)Dr * Ds * 2);
  __bf16* soT = (__bf16*)alloc((size_t)Ds * Ds * 2);
  __bf16* rqT = (__bf16*)alloc((size_t)Dr * Dr * 2);
  __bf16* rkT = (__bf16*)alloc((size_t)Ds * Dr * 2);
  __bf16* rvT = (__bf16*)alloc((size_t)Ds * Dr * 2);
  __bf16* roT = (__bf16*)alloc((size_t)Dr * Dr * 2);

  dim3 blk(256, 1, 1);
  auto cdiv = [](long long a, long long b) { return (unsigned)((a + b - 1) / b); };

  // --- weights -> bf16 transposed ---
  wcast_kernel<<<cdiv((long long)Ds * Ds, 256), blk, 0, stream>>>(sq, sqT, Ds, Ds);
  wcast_kernel<<<cdiv((long long)Dr * Ds, 256), blk, 0, stream>>>(sk, skT, Dr, Ds);
  wcast_kernel<<<cdiv((long long)Dr * Ds, 256), blk, 0, stream>>>(sv, svT, Dr, Ds);
  wcast_kernel<<<cdiv((long long)Ds * Ds, 256), blk, 0, stream>>>(so, soT, Ds, Ds);
  wcast_kernel<<<cdiv((long long)Dr * Dr, 256), blk, 0, stream>>>(rq, rqT, Dr, Dr);
  wcast_kernel<<<cdiv((long long)Ds * Dr, 256), blk, 0, stream>>>(rk, rkT, Ds, Dr);
  wcast_kernel<<<cdiv((long long)Ds * Dr, 256), blk, 0, stream>>>(rv, rvT, Ds, Dr);
  wcast_kernel<<<cdiv((long long)Dr * Dr, 256), blk, 0, stream>>>(ro, roT, Dr, Dr);

  // --- RMSNorm ---
  rmsnorm_kernel<<<B * T, blk, 0, stream>>>(s, nsw, sn, Ds);
  rmsnorm_kernel<<<B * Nn, blk, 0, stream>>>(r, nrw, rn, Dr);

  // --- projections (bf16 WMMA) ---
  gemm_wmma_kernel<2, false, true, false, false><<<dim3(Ds / 64, (B * T) / 256, 1), blk, 0, stream>>>(
      sn, Ds, 0, 0, sqT, Ds, 0, 0, qs, Ds, 0, 0, nullptr, 0, 0, Ds, 1, 1.f);
  gemm_wmma_kernel<1, false, true, false, false><<<dim3(Ds / 64, (B * Nn) / 128, 1), blk, 0, stream>>>(
      rn, Dr, 0, 0, skT, Dr, 0, 0, ks, Ds, 0, 0, nullptr, 0, 0, Dr, 1, 1.f);
  gemm_wmma_kernel<1, false, true, false, false><<<dim3(Ds / 64, (B * Nn) / 128, 1), blk, 0, stream>>>(
      rn, Dr, 0, 0, svT, Dr, 0, 0, vs, Ds, 0, 0, nullptr, 0, 0, Dr, 1, 1.f);
  gemm_wmma_kernel<1, false, true, true, false><<<dim3(Ds / 64, 1, B), blk, 0, stream>>>(
      rn, Dr, (long long)Nn * Dr, 0, svT, Dr, 0, 0,
      vts, Nn, (long long)Ds * Nn, 0, nullptr, 0, 0, Dr, 1, 1.f);
  gemm_wmma_kernel<1, false, true, false, false><<<dim3(Dr / 64, (B * Nn) / 128, 1), blk, 0, stream>>>(
      rn, Dr, 0, 0, rqT, Dr, 0, 0, qrp, Dr, 0, 0, nullptr, 0, 0, Dr, 1, 1.f);
  gemm_wmma_kernel<2, false, true, false, false><<<dim3(Dr / 64, (B * T) / 256, 1), blk, 0, stream>>>(
      sn, Ds, 0, 0, rkT, Ds, 0, 0, krp, Dr, 0, 0, nullptr, 0, 0, Ds, 1, 1.f);
  gemm_wmma_kernel<2, false, true, true, false><<<dim3(Dr / 64, T / 256, B), blk, 0, stream>>>(
      sn, Ds, (long long)T * Ds, 0, rvT, Ds, 0, 0,
      vrt, T, (long long)Dr * T, 0, nullptr, 0, 0, Ds, 1, 1.f);

  // --- attention 1: S reads R (softmax over N=128) ---
  gemm_wmma_kernel<2, false, false, false, false><<<dim3(Nn / 64, T / 256, B * H), blk, 0, stream>>>(
      qs, Ds, (long long)T * Ds, dhs, ks, Ds, (long long)Nn * Ds, dhs,
      sc, Nn, (long long)H * T * Nn, (long long)T * Nn, nullptr, 0, 0, dhs, H, scale_s);
  softmax128_kernel<<<cdiv((long long)B * H * T * 32, 256), blk, 0, stream>>>(sc, (long long)B * H * T);
  gemm_wmma_kernel<2, true, true, false, false><<<dim3(dhs / 64, T / 256, B * H), blk, 0, stream>>>(
      sc, Nn, (long long)H * T * Nn, (long long)T * Nn,
      vts, Nn, (long long)Ds * Nn, (long long)dhs * Nn,
      a1, Ds, (long long)T * Ds, dhs, nullptr, 0, 0, Nn, H, 1.f);
  gemm_wmma_kernel<2, false, false, false, true><<<dim3(Ds / 64, (B * T) / 256, 1), blk, 0, stream>>>(
      a1, Ds, 0, 0, soT, Ds, 0, 0, s_out, Ds, 0, 0, s, 0, 0, Ds, 1, 1.f);

  // --- attention 2: R reads S (softmax over T=4096) ---
  gemm_wmma_kernel<1, false, false, false, false><<<dim3(T / 64, Nn / 128, B * H), blk, 0, stream>>>(
      qrp, Dr, (long long)Nn * Dr, dhr, krp, Dr, (long long)T * Dr, dhr,
      sc, T, (long long)H * Nn * T, (long long)Nn * T, nullptr, 0, 0, dhr, H, scale_r);
  softmax4096_kernel<<<B * H * Nn, blk, 0, stream>>>(sc);
  gemm_wmma_kernel<1, true, true, false, false><<<dim3(dhr / 64, Nn / 128, B * H), blk, 0, stream>>>(
      sc, T, (long long)H * Nn * T, (long long)Nn * T,
      vrt, T, (long long)Dr * T, (long long)dhr * T,
      a2, Dr, (long long)Nn * Dr, dhr, nullptr, 0, 0, T, H, 1.f);
  gemm_wmma_kernel<1, false, false, false, true><<<dim3(Dr / 64, (B * Nn) / 128, 1), blk, 0, stream>>>(
      a2, Dr, 0, 0, roT, Dr, 0, 0, r_out, Dr, 0, 0, r, 0, 0, Dr, 1, 1.f);

  // --- sparse top-k path: sc2 = q_s @ k_s^T over full Ds, then top-8 ---
  gemm_wmma_kernel<2, false, false, false, false><<<dim3(Nn / 64, T / 256, B), blk, 0, stream>>>(
      qs, Ds, (long long)T * Ds, 0, ks, Ds, (long long)Nn * Ds, 0,
      sc, Nn, (long long)T * Nn, 0, nullptr, 0, 0, Ds, 1, scale_s);
  sparse_topk_kernel<<<(B * T * 32) / 256, blk, 0, stream>>>(sc, vs, gate, s_out);
}
